// SAGE_76287209112087
// MI455X (gfx1250) — matrix operand order repaired and checked by reading, not compiled, not run on hardware
//
#include <hip/hip_runtime.h>

#define N_NODES 50000
#define N_EDGES 800000
#define F_IN    128
#define F_H     256
#define N_CLS   47
#define CPAD    48   // class dim padded to 3 WMMA tiles

typedef __attribute__((ext_vector_type(16))) __bf16 v16bf;
typedef __attribute__((ext_vector_type(8)))  float  v8f;

// float -> bf16 bits, round-to-nearest-even (used only in weight prep)
__device__ __forceinline__ unsigned short f2bf(float f) {
  union { float f; unsigned u; } v; v.f = f;
  unsigned u = v.u;
  unsigned r = u + 0x7FFFu + ((u >> 16) & 1u);
  return (unsigned short)(r >> 16);
}

__global__ void k_zero(float* p, long n) {
  long i = (long)blockIdx.x * blockDim.x + threadIdx.x;
  long s = (long)gridDim.x * blockDim.x;
  for (; i < n; i += s) p[i] = 0.f;
}

// Build transposed bf16 weight buffers:
// B1t[n][k], n in [0,256), k in [0,256): k<128 -> w1n[k][n], else w1s[k-128][n]
// B2t[n][k], n in [0,48),  k in [0,512): n<47 ? (k<256 ? w2n[k][n] : w2s[k-256][n]) : 0
__global__ void k_prep_w(const float* __restrict__ w1n, const float* __restrict__ w1s,
                         const float* __restrict__ w2n, const float* __restrict__ w2s,
                         unsigned short* __restrict__ B1t, unsigned short* __restrict__ B2t) {
  int tid = blockIdx.x * blockDim.x + threadIdx.x;
  const int n1 = 256 * 256;
  const int n2 = CPAD * 512;
  if (tid < n1) {
    int n = tid >> 8, k = tid & 255;
    float v = (k < 128) ? w1n[k * 256 + n] : w1s[(k - 128) * 256 + n];
    B1t[n * 256 + k] = f2bf(v);
  } else if (tid < n1 + n2) {
    int t = tid - n1;
    int n = t >> 9, k = t & 511;
    float v = 0.f;
    if (n < N_CLS) v = (k < 256) ? w2n[k * N_CLS + n] : w2s[(k - 256) * N_CLS + n];
    B2t[n * 512 + k] = f2bf(v);
  }
}

__global__ void k_degree(const int* __restrict__ row, float* __restrict__ deg) {
  int e = blockIdx.x * blockDim.x + threadIdx.x;
  if (e < N_EDGES) unsafeAtomicAdd(&deg[row[e]], 1.f);
}

// One thread per (edge, 4-float chunk): float4 gather from src[col], 4 atomics into dst[row]
template<int W>
__global__ void k_scatter(const int* __restrict__ row, const int* __restrict__ col,
                          const float* __restrict__ src, float* __restrict__ dst) {
  const int Q = W / 4;
  long tid = (long)blockIdx.x * blockDim.x + threadIdx.x;
  long e = tid / Q;
  int  q = (int)(tid % Q);
  if (e >= N_EDGES) return;
  const float4 v = ((const float4*)(src + (long)col[e] * W))[q];
  float* d = dst + (long)row[e] * W + q * 4;
  unsafeAtomicAdd(d + 0, v.x);
  unsafeAtomicAdd(d + 1, v.y);
  unsafeAtomicAdd(d + 2, v.z);
  unsafeAtomicAdd(d + 3, v.w);
}

template<int W>
__global__ void k_divdeg(float* __restrict__ a, const float* __restrict__ deg) {
  long i = (long)blockIdx.x * blockDim.x + threadIdx.x;
  if (i >= (long)N_NODES * W) return;
  float d = deg[i / W];
  a[i] = (d > 0.f) ? a[i] / d : 0.f;
}

__device__ __forceinline__ v16bf cvt16(float4 q0, float4 q1, float4 q2, float4 q3) {
  v16bf a = { (__bf16)q0.x, (__bf16)q0.y, (__bf16)q0.z, (__bf16)q0.w,
              (__bf16)q1.x, (__bf16)q1.y, (__bf16)q1.z, (__bf16)q1.w,
              (__bf16)q2.x, (__bf16)q2.y, (__bf16)q2.z, (__bf16)q2.w,
              (__bf16)q3.x, (__bf16)q3.y, (__bf16)q3.z, (__bf16)q3.w };
  return a;
}

union BFu { v16bf v; uint4 q[2]; };

// One wave computes NT adjacent 16x16 output tiles (shares the A fragment).
// A = [A0 | A1] row-major f32, widths KT0*32 / KT1*32, converted to bf16 in-register.
// Bt = [nOut][Ktot] bf16 (pre-transposed -> B fragment is two 8-bf16 runs, same k order as A).
// KT0/KT1/NT/NG compile-time so the K loop fully unrolls (KT0+KT1 steps, NT WMMAs each).
template<int KT0, int KT1, int NT, int NG>
__global__ void k_gemm_wmma(const float* __restrict__ A0, const float* __restrict__ A1,
                            const unsigned short* __restrict__ Bt,
                            const float* __restrict__ bias0, const float* __restrict__ bias1,
                            int blen, float* __restrict__ Cm, int ldc, int do_relu) {
  const int Ktot = (KT0 + KT1) * 32;
  int lane = threadIdx.x & 31;
  int wid  = (blockIdx.x * blockDim.x + threadIdx.x) >> 5;
  const int nWaves = 3125 * NG;                 // N/16 row tiles * column-tile groups
  if (wid >= nWaves) return;                    // whole wave exits; EXEC all-1s for WMMA
  int mtile = wid / NG, g = wid % NG;
  int m = lane & 15, half = lane >> 4;
  long rowA = (long)mtile * 16 + m;             // A row owned by this lane
  const float* a0p = A0 + rowA * (KT0 * 32);
  const float* a1p = A1 + rowA * (KT1 * 32);

  int ncol[NT];
  const unsigned short* bp[NT];
#pragma unroll
  for (int t = 0; t < NT; t++) {
    ncol[t] = (g * NT + t) * 16 + m;            // output column owned by this lane
    bp[t]   = Bt + (long)ncol[t] * Ktot + (half << 3);
  }

  v8f c[NT];
#pragma unroll
  for (int t = 0; t < NT; t++) c[t] = (v8f){};

#pragma unroll
  for (int s = 0; s < KT0 + KT1; s++) {
    // compile-time source select after unroll
    const float* sp = (s < KT0) ? (a0p + s * 32) : (a1p + (s - KT0) * 32);
    // A 16x32 bf16 layout: lane half 0 -> K {0..7,16..23}, half 1 -> +8.
    // Two contiguous 8-float ranges per lane -> 4x b128 loads, then packed cvt.
    float4 q0 = *(const float4*)(sp + (half << 3));
    float4 q1 = *(const float4*)(sp + (half << 3) + 4);
    float4 q2 = *(const float4*)(sp + 16 + (half << 3));
    float4 q3 = *(const float4*)(sp + 20 + (half << 3));
    v16bf a = cvt16(q0, q1, q2, q3);
#pragma unroll
    for (int t = 0; t < NT; t++) {
      BFu b;                                     // two 8-bf16 runs (k 8h.., 16+8h..) -> 2x b128 loads
      b.q[0] = *(const uint4*)(bp[t] + s * 32);
      b.q[1] = *(const uint4*)(bp[t] + s * 32 + 16);
      c[t] = __builtin_amdgcn_wmma_f32_16x16x32_bf16(false, a, false, b.v,
                                                     (short)0, c[t], false, false);
    }
  }

#pragma unroll
  for (int t = 0; t < NT; t++) {
    float bias = (ncol[t] < blen) ? (bias0[ncol[t]] + bias1[ncol[t]]) : 0.f;
#pragma unroll
    for (int j = 0; j < 8; j++) {                // C/D: vgpr j -> row m = j + 8*half
      long mr = (long)mtile * 16 + j + (half << 3);
      float v = c[t][j] + bias;
      if (do_relu) v = fmaxf(v, 0.f);
      Cm[mr * ldc + ncol[t]] = v;
    }
  }
}

__global__ void k_loss(const float* __restrict__ logits, const int* __restrict__ y,
                       const unsigned char* __restrict__ mask, float* __restrict__ scal) {
  __shared__ float sl[256], sc[256];
  int i = blockIdx.x * blockDim.x + threadIdx.x;
  float nll = 0.f, cnt = 0.f;
  if (i < N_NODES && mask[i]) {
    const float* lg = logits + (long)i * CPAD;
    float mx = lg[0];
    for (int c = 1; c < N_CLS; c++) mx = fmaxf(mx, lg[c]);
    float se = 0.f;
    for (int c = 0; c < N_CLS; c++) se += __expf(lg[c] - mx);
    float lp = lg[y[i]] - mx - __logf(se);
    nll = -lp; cnt = 1.f;
  }
  sl[threadIdx.x] = nll; sc[threadIdx.x] = cnt;
  __syncthreads();
  for (int o = 128; o > 0; o >>= 1) {
    if (threadIdx.x < o) { sl[threadIdx.x] += sl[threadIdx.x + o]; sc[threadIdx.x] += sc[threadIdx.x + o]; }
    __syncthreads();
  }
  if (threadIdx.x == 0) {
    unsafeAtomicAdd(&scal[0], sl[0]);
    unsafeAtomicAdd(&scal[1], sc[0]);
  }
}

__global__ void k_final(const float* __restrict__ scal, float* __restrict__ out) {
  if (blockIdx.x == 0 && threadIdx.x == 0)
    out[0] = scal[0] / fmaxf(scal[1], 1.f);
}

extern "C" void kernel_launch(void* const* d_in, const int* in_sizes, int n_in,
                              void* d_out, int out_size, void* d_ws, size_t ws_size,
                              hipStream_t stream) {
  const float* x   = (const float*)d_in[0];
  const int*   row = (const int*)d_in[1];
  const int*   col = (const int*)d_in[2];
  const int*   y   = (const int*)d_in[3];
  const unsigned char* mask = (const unsigned char*)d_in[4];  // jnp.bool_ -> 1 byte
  const float* w1n = (const float*)d_in[5];
  const float* b1n = (const float*)d_in[6];
  const float* w1s = (const float*)d_in[7];
  const float* b1s = (const float*)d_in[8];
  const float* w2n = (const float*)d_in[9];
  const float* b2n = (const float*)d_in[10];
  const float* w2s = (const float*)d_in[11];
  const float* b2s = (const float*)d_in[12];

  // Workspace layout: the zeroed region [scal|deg|agg1|agg2] is contiguous.
  float* scal = (float*)d_ws;                         // 16 floats: [0]=loss sum [1]=count
  float* deg  = scal + 16;                            // N
  float* agg1 = deg  + N_NODES;                       // N*128
  float* agg2 = agg1 + (long)N_NODES * F_IN;          // N*256
  float* h1   = agg2 + (long)N_NODES * F_H;           // N*256 (fully overwritten)
  float* lgts = h1   + (long)N_NODES * F_H;           // N*48  (fully overwritten)
  unsigned short* B1t = (unsigned short*)(lgts + (long)N_NODES * CPAD); // 256*256 bf16 (16B-aligned)
  unsigned short* B2t = B1t + 256 * 256;                                // 48*512 bf16 (16B-aligned)

  long zn = 16 + N_NODES + (long)N_NODES * F_IN + (long)N_NODES * F_H;
  k_zero<<<4096, 256, 0, stream>>>(scal, zn);

  k_prep_w<<<(256 * 256 + CPAD * 512 + 255) / 256, 256, 0, stream>>>(w1n, w1s, w2n, w2s, B1t, B2t);

  k_degree<<<(N_EDGES + 255) / 256, 256, 0, stream>>>(row, deg);

  { long t = (long)N_EDGES * (F_IN / 4);
    k_scatter<F_IN><<<(int)((t + 255) / 256), 256, 0, stream>>>(row, col, x, agg1); }
  { long t = (long)N_NODES * F_IN;
    k_divdeg<F_IN><<<(int)((t + 255) / 256), 256, 0, stream>>>(agg1, deg); }

  // h1 = relu([agg1|x] @ [w1n;w1s] + b1n + b1s): K=256 (4+4 k-steps),
  // 16 col tiles handled as 4 groups of NT=4 per wave.
  { int w = 3125 * 4;
    k_gemm_wmma<4, 4, 4, 4><<<(w + 7) / 8, 256, 0, stream>>>(
        agg1, x, B1t, b1n, b1s, 256, h1, F_H, 1); }

  { long t = (long)N_EDGES * (F_H / 4);
    k_scatter<F_H><<<(int)((t + 255) / 256), 256, 0, stream>>>(row, col, h1, agg2); }
  { long t = (long)N_NODES * F_H;
    k_divdeg<F_H><<<(int)((t + 255) / 256), 256, 0, stream>>>(agg2, deg); }

  // logits = [agg2|h1] @ [w2n;w2s] + b2n + b2s: K=512 (8+8 k-steps),
  // 3 col tiles (48 padded), all NT=3 in one group per wave.
  { int w = 3125 * 1;
    k_gemm_wmma<8, 8, 3, 1><<<(w + 7) / 8, 256, 0, stream>>>(
        agg2, h1, B2t, b2n, b2s, N_CLS, lgts, CPAD, 0); }

  k_loss<<<(N_NODES + 255) / 256, 256, 0, stream>>>(lgts, y, mask, scal);
  k_final<<<1, 64, 0, stream>>>(scal, (float*)d_out);
}
